// LieDeformableConv2D_67070209295035
// MI455X (gfx1250) — matrix operand order repaired and checked by reading, not compiled, not run on hardware
//
#include <hip/hip_runtime.h>
#include <hip/hip_bf16.h>
#include <math.h>

typedef __attribute__((ext_vector_type(16))) _Float16 v16h;
typedef __attribute__((ext_vector_type(8)))  float    v8f;
typedef __attribute__((ext_vector_type(4)))  float    v4f;
typedef __attribute__((ext_vector_type(4)))  unsigned int v4u;

#define BATCH 4
#define CIN   64
#define COUT  64
#define Hh    128
#define Ww    128
#define HW    (Hh*Ww)        // 16384
#define NTAP  9
#define KTOT  (CIN*NTAP)     // 576
#define NPIX  (BATCH*HW)     // 65536

// ---------------------------------------------------------------------------
// Kernel T: NCHW -> NHWC transpose of x (channels innermost) via LDS tiles.
// xT[b][p][c] = x[b][c][p],  p = h*W+w.  Coalesced reads and writes.
// ---------------------------------------------------------------------------
__global__ void transpose_nchw_nhwc_kernel(const float* __restrict__ x,
                                           float* __restrict__ xT)
{
    __shared__ float tile[CIN][65];          // 64x64 tile, pad to kill conflicts
    const int blk = blockIdx.x;              // b * 256 + pixel-tile
    const int b   = blk >> 8;
    const int p0  = (blk & 255) * 64;        // first pixel of tile
    const float* xb = x  + (size_t)b * CIN * HW;
    float*       ob = xT + (size_t)b * HW * CIN;

    {
        const int tp = threadIdx.x & 63;
        const int c0 = threadIdx.x >> 6;     // 0..3
        #pragma unroll
        for (int c = c0; c < CIN; c += 4)
            tile[c][tp] = xb[(size_t)c * HW + p0 + tp];
    }
    __syncthreads();
    {
        const int tc = threadIdx.x & 63;
        const int r0 = threadIdx.x >> 6;     // 0..3
        #pragma unroll
        for (int tp = r0; tp < 64; tp += 4)
            ob[(size_t)(p0 + tp) * CIN + tc] = tile[tc][tp];
    }
}

// ---------------------------------------------------------------------------
// Kernel W: one-time weight prep. Builds the permuted half-precision B image
//   wBg[o][t*64 + c] = w_conv[o][c][t]; channels kept in natural order
//   (ch = s, no swap), matching the contiguous channels A loads.
// Done ONCE; every GEMM block then bulk-copies this 72KB image into LDS.
// ---------------------------------------------------------------------------
__global__ void weight_prep_kernel(const float* __restrict__ w_conv,
                                   _Float16* __restrict__ wBg)
{
    int i = blockIdx.x * blockDim.x + threadIdx.x;   // 0 .. COUT*KTOT-1
    if (i >= COUT * KTOT) return;
    int o  = i / KTOT;
    int k  = i - o * KTOT;
    int t  = k >> 6;
    int cc = k & 63;
    int s  = cc & 31;
    int g  = s >> 3;
    int ch = s;
    int c  = (cc & 32) + ch;
    wBg[i] = (_Float16)w_conv[(o * CIN + c) * NTAP + t];
}

// ---------------------------------------------------------------------------
// Kernel A: offset conv + Lie homography -> absolute sample coords (py,px)
// Reads NHWC xT; per-pixel 9 neighbors x 64 contiguous channels (v4f dots).
// ---------------------------------------------------------------------------
__global__ void lie_offsets_kernel(const float* __restrict__ xT,
                                   const float* __restrict__ w_off,
                                   const float* __restrict__ P,
                                   const float* __restrict__ d,
                                   const float* __restrict__ Pinv,
                                   float* __restrict__ pypx)
{
    __shared__ v4f swot4[NTAP * (CIN / 4)];  // weights re-staged [tap][c]
    __shared__ float sP[9], sPi[9], sLd[3];
    float* swot = (float*)swot4;
    for (int i = threadIdx.x; i < NTAP * CIN; i += blockDim.x) {
        int t = i >> 6, c = i & 63;
        swot[t * CIN + c] = w_off[c * NTAP + t];
    }
    if (threadIdx.x < 9)        sP[threadIdx.x]       = P[threadIdx.x];
    else if (threadIdx.x < 18)  sPi[threadIdx.x - 9]  = Pinv[threadIdx.x - 9];
    else if (threadIdx.x < 21)  sLd[threadIdx.x - 18] = __logf(d[threadIdx.x - 18]);
    __syncthreads();

    int p = blockIdx.x * blockDim.x + threadIdx.x;
    int w = p & (Ww - 1);
    int h = (p >> 7) & (Hh - 1);
    int b = p >> 14;
    const size_t bHW = (size_t)b * HW;

    v4f acc4 = {};
    #pragma unroll
    for (int t = 0; t < NTAP; ++t) {
        int yy = h + (t / 3) - 1;
        int xx = w + (t % 3) - 1;
        if ((unsigned)yy >= (unsigned)Hh || (unsigned)xx >= (unsigned)Ww) continue;
        const v4f* q  = (const v4f*)(xT + (bHW + yy * Ww + xx) * CIN);
        const v4f* wv = (const v4f*)(swot + t * CIN);
        #pragma unroll
        for (int g = 0; g < CIN / 4; ++g) acc4 += q[g] * wv[g];
    }
    float raw = acc4[0] + acc4[1] + acc4[2] + acc4[3];

    float lw  = fminf(fmaxf(raw + 0.5f, 0.f), 1.f);
    float pw0 = __expf(lw * sLd[0]);
    float pw1 = __expf(lw * sLd[1]);
    float pw2 = __expf(lw * sLd[2]);

    float Hm[9];
    #pragma unroll
    for (int i = 0; i < 3; ++i)
        #pragma unroll
        for (int j = 0; j < 3; ++j)
            Hm[i * 3 + j] = sP[i * 3 + 0] * pw0 * sPi[0 * 3 + j]
                          + sP[i * 3 + 1] * pw1 * sPi[1 * 3 + j]
                          + sP[i * 3 + 2] * pw2 * sPi[2 * 3 + j];

    float* op = pypx + (size_t)p * (2 * NTAP);
    #pragma unroll
    for (int t = 0; t < NTAP; ++t) {
        float ptx = (float)(t % 3) - 1.f;
        float pty = (float)(t / 3) - 1.f;
        float qx = Hm[0] * ptx + Hm[1] * pty + Hm[2];
        float qy = Hm[3] * ptx + Hm[4] * pty + Hm[5];
        float qz = Hm[6] * ptx + Hm[7] * pty + Hm[8];
        float rz = __frcp_rn(qz);
        op[2 * t + 0] = (float)h + qy * rz;   // py = h + qy/qz
        op[2 * t + 1] = (float)w + qx * rz;   // px = w + qx/qz
    }
}

// ---------------------------------------------------------------------------
// Kernel B: fused bilinear gather + WMMA GEMM (NHWC input).
// Block = 256 threads = 8 waves; wave = 16 pixels x 64 outputs (4x v8f acc).
// K = 576 = 9 taps x 2 chunks of 32 channels via v_wmma_f32_16x16x32_f16.
// LDS staging is a bulk b128 copy of the pre-permuted 72KB weight image.
// ---------------------------------------------------------------------------
__global__ void lie_deform_wmma_kernel(const float* __restrict__ xT,
                                       const _Float16* __restrict__ wBg,
                                       const float* __restrict__ pypx,
                                       float* __restrict__ out)
{
    extern __shared__ _Float16 wB[];   // [COUT][KTOT] = 73728 B

    // bulk vectorized staging: 4608 x 16B, 18 per thread
    {
        const v4u* src = (const v4u*)wBg;
        v4u*       dst = (v4u*)wB;
        #pragma unroll
        for (int i = threadIdx.x; i < (COUT * KTOT) / 8; i += 256)
            dst[i] = src[i];
    }
    __syncthreads();

    const int lane  = threadIdx.x & 31;
    const int wave  = threadIdx.x >> 5;
    const int r     = lane & 15;     // A row (pixel within tile)
    const int hi    = lane >> 4;     // lane-half -> channel half within chunk
    const int mbase = blockIdx.x * 128 + wave * 16;
    const int p     = mbase + r;
    const int pb    = p >> 14;
    const size_t rowbase = (size_t)pb * HW;
    const float* cp = pypx + (size_t)p * (2 * NTAP);

    v8f acc0 = {}, acc1 = {}, acc2 = {}, acc3 = {};

    for (int t = 0; t < NTAP; ++t) {
        float py = cp[2 * t + 0];
        float px = cp[2 * t + 1];
        float y0f = floorf(py), x0f = floorf(px);
        float wy1 = py - y0f, wy0 = 1.f - wy1;
        float wx1 = px - x0f, wx0 = 1.f - wx1;
        int y0 = (int)y0f, x0 = (int)x0f;
        int y1 = y0 + 1,  x1 = x0 + 1;
        bool y0ok = (y0 >= 0) & (y0 <= Hh - 1);
        bool y1ok = (y1 >= 0) & (y1 <= Hh - 1);
        bool x0ok = (x0 >= 0) & (x0 <= Ww - 1);
        bool x1ok = (x1 >= 0) & (x1 <= Ww - 1);
        int y0c = min(max(y0, 0), Hh - 1), y1c = min(max(y1, 0), Hh - 1);
        int x0c = min(max(x0, 0), Ww - 1), x1c = min(max(x1, 0), Ww - 1);
        float v00 = (y0ok & x0ok) ? wy0 * wx0 : 0.f;
        float v01 = (y0ok & x1ok) ? wy0 * wx1 : 0.f;
        float v10 = (y1ok & x0ok) ? wy1 * wx0 : 0.f;
        float v11 = (y1ok & x1ok) ? wy1 * wx1 : 0.f;
        const float* c00 = xT + (rowbase + y0c * Ww + x0c) * CIN;
        const float* c01 = xT + (rowbase + y0c * Ww + x1c) * CIN;
        const float* c10 = xT + (rowbase + y1c * Ww + x0c) * CIN;
        const float* c11 = xT + (rowbase + y1c * Ww + x1c) * CIN;

        #pragma unroll
        for (int cb = 0; cb < CIN; cb += 32) {
            const int cbase = cb + hi * 16;      // 16 contiguous channels
            const v4f* q00 = (const v4f*)(c00 + cbase);
            const v4f* q01 = (const v4f*)(c01 + cbase);
            const v4f* q10 = (const v4f*)(c10 + cbase);
            const v4f* q11 = (const v4f*)(c11 + cbase);

            v16h a;
            #pragma unroll
            for (int g = 0; g < 4; ++g) {
                v4f s = v00 * q00[g] + v01 * q01[g] + v10 * q10[g] + v11 * q11[g];
                a[4 * g + 0] = (_Float16)s[0];
                a[4 * g + 1] = (_Float16)s[1];
                a[4 * g + 2] = (_Float16)s[2];
                a[4 * g + 3] = (_Float16)s[3];
            }

            // B fragments: lanes 0-15 K-slots 0..15, lanes 16-31 slots 16..31
            const int kB = t * 64 + cb + hi * 16;   // 16 halves, 32B aligned
            const int oc = lane & 15;
            v16h b0 = *(const v16h*)&wB[(0 * 16 + oc) * KTOT + kB];
            v16h b1 = *(const v16h*)&wB[(1 * 16 + oc) * KTOT + kB];
            v16h b2 = *(const v16h*)&wB[(2 * 16 + oc) * KTOT + kB];
            v16h b3 = *(const v16h*)&wB[(3 * 16 + oc) * KTOT + kB];

            acc0 = __builtin_amdgcn_wmma_f32_16x16x32_f16(false, a, false, b0, (short)0, acc0, false, false);
            acc1 = __builtin_amdgcn_wmma_f32_16x16x32_f16(false, a, false, b1, (short)0, acc1, false, false);
            acc2 = __builtin_amdgcn_wmma_f32_16x16x32_f16(false, a, false, b2, (short)0, acc2, false, false);
            acc3 = __builtin_amdgcn_wmma_f32_16x16x32_f16(false, a, false, b3, (short)0, acc3, false, false);
        }
    }

    // store D: VGPR j -> row M = j + 8*hi, col N = lane&15 (NCHW output).
    // Rows map to consecutive w -> compiler vectorizes to global_store_b128.
    const int oc = lane & 15;
    #pragma unroll
    for (int j = 0; j < 8; ++j) {
        int m  = j + 8 * hi;
        int pp = mbase + m;
        int sw = pp & (Ww - 1);
        int sh = (pp >> 7) & (Hh - 1);
        int sb = pp >> 14;
        float* ob = out + (size_t)sb * COUT * HW + sh * Ww + sw;
        ob[(0 * 16 + oc) * HW] = acc0[j];
        ob[(1 * 16 + oc) * HW] = acc1[j];
        ob[(2 * 16 + oc) * HW] = acc2[j];
        ob[(3 * 16 + oc) * HW] = acc3[j];
    }
}

// ---------------------------------------------------------------------------
extern "C" void kernel_launch(void* const* d_in, const int* in_sizes, int n_in,
                              void* d_out, int out_size, void* d_ws, size_t ws_size,
                              hipStream_t stream) {
    const float* x      = (const float*)d_in[0];
    const float* w_off  = (const float*)d_in[1];
    const float* w_conv = (const float*)d_in[2];
    const float* P      = (const float*)d_in[3];
    const float* d      = (const float*)d_in[4];
    const float* Pinv   = (const float*)d_in[5];
    float* out  = (float*)d_out;

    float*    xT   = (float*)d_ws;                          // 16.8 MB
    float*    pypx = xT + (size_t)BATCH * HW * CIN;         //  4.7 MB
    _Float16* wBg  = (_Float16*)(pypx + (size_t)NPIX * 2 * NTAP);  // 72 KB

    // 1) NCHW -> NHWC (channels innermost) for vectorized gathers
    transpose_nchw_nhwc_kernel<<<BATCH * (HW / 64), 256, 0, stream>>>(x, xT);

    // 2) one-time permuted f16 weight image (72 KB, L2-resident)
    weight_prep_kernel<<<(COUT * KTOT + 255) / 256, 256, 0, stream>>>(w_conv, wBg);

    // 3) per-pixel Lie offsets -> absolute sample coords
    lie_offsets_kernel<<<NPIX / 256, 256, 0, stream>>>(xT, w_off, P, d, Pinv, pypx);

    // 4) fused bilinear gather + WMMA GEMM; 72 KB dynamic LDS for weights
    const size_t ldsB = (size_t)COUT * KTOT * sizeof(_Float16);  // 73728 B
    lie_deform_wmma_kernel<<<NPIX / 128, 256, ldsB, stream>>>(xT, wBg, pypx, out);
}
